// MainModel_83940840833168
// MI455X (gfx1250) — matrix operand order repaired and checked by reading, not compiled, not run on hardware
//
#include <hip/hip_runtime.h>

// ---------------- problem constants ----------------
constexpr int Bn  = 32;     // batch
constexpr int C   = 3072;   // channels
constexpr int HWD = 784;    // 28*28
constexpr int K   = 11;     // concepts
constexpr int KP  = 16;     // padded concepts (WMMA M/N tile)
constexpr int NC  = 200;    // classes

typedef __attribute__((ext_vector_type(16))) __bf16 v16bf;
typedef __attribute__((ext_vector_type(8)))  float  v8f;
typedef unsigned int u32x4 __attribute__((ext_vector_type(4)));
typedef unsigned int u32x8 __attribute__((ext_vector_type(8)));

struct U32x8s { uint4 a, b; };   // 32-byte carrier for bit_cast to v16bf

static __device__ __forceinline__ unsigned short f2bfu(float f) {
    unsigned u = __builtin_bit_cast(unsigned, f);
    return (unsigned short)((u + 0x7FFFu + ((u >> 16) & 1u)) >> 16);
}
static __device__ __forceinline__ __bf16 f2bf(float f) {
    return __builtin_bit_cast(__bf16, f2bfu(f));
}
static __device__ __forceinline__ float bflo(unsigned p) {   // low bf16 of packed u32
    return __builtin_bit_cast(float, p << 16);
}
static __device__ __forceinline__ float bfhi(unsigned p) {   // high bf16 of packed u32
    return __builtin_bit_cast(float, p & 0xFFFF0000u);
}

// ---------------- kernel 0: init ----------------
__global__ void init_kernel(const float* __restrict__ w_land,
                            float* __restrict__ attn_out,
                            __bf16* __restrict__ wland_bf,
                            float* __restrict__ a_sq) {
    int tid = blockIdx.x * blockDim.x + threadIdx.x;
    int nt  = gridDim.x * blockDim.x;
    if (tid < Bn * K) attn_out[tid] = 0.0f;
    for (int i = tid; i < KP * C; i += nt) {
        int k = i / C, c = i - k * C;
        float v = (k < K) ? w_land[k * C + c] : 0.0f;
        wland_bf[i] = f2bf(v);
    }
    if (tid < K) {
        float s = 0.0f;
        for (int c = 0; c < C; ++c) { float w = w_land[tid * C + c]; s += w * w; }
        a_sq[tid] = s;
    }
}

// ---------------- kernel 1: distances -> softmax maps + attn_cls ----------------
// grid = Bn*13, block = 128 (4 waves), each wave owns a 16-pixel output tile.
// LDS x-tile is stored TRANSPOSED [pixel][chan] in bf16 (row stride 40 halves =
// 80B, 16B aligned) so B-fragments are 2x ds_load_b128 per WMMA.
__global__ __launch_bounds__(128)
void maps_kernel(const float* __restrict__ x,
                 const __bf16* __restrict__ wland_bf,
                 const float* __restrict__ a_sq_g,
                 float* __restrict__ maps_out,
                 float* __restrict__ attn_out) {
    const int b   = blockIdx.x / 13;
    const int cb  = blockIdx.x % 13;
    const int hw0 = cb * 64;
    const int tid  = threadIdx.x;
    const int lane = tid & 31;
    const int wave = tid >> 5;

    __shared__ __align__(16) __bf16 lds_xT[64][40];  // [pixel][chan-chunk]
    __shared__ float sm_ab[64][KP];
    __shared__ float sattn[KP];
    if (tid < KP) sattn[tid] = 0.0f;

    const bool tileValid = (cb * 4 + wave) < (HWD / 16);  // 49 pixel-tiles
    const int lm    = lane & 15;
    const int lhalf = lane >> 4;
    const int abase = lhalf ? 8 : 0;    // A-matrix K-chunk base (ISA 16-bit A layout)
    const int rb    = lhalf ? 8 : 0;    // B-matrix row base (ISA 16-bit B layout)
    const int lcol  = wave * 16 + lm;

    v8f   acc = {};
    float bsq = 0.0f;

    for (int c0 = 0; c0 < C; c0 += 32) {
        __syncthreads();  // protect previous chunk's LDS reads
        // stage: 4 channel rows per thread per step, packed bf16x4 -> ds_store_b64
        #pragma unroll
        for (int i = 0; i < 4; ++i) {
            int e   = tid + i * 128;
            int col = e & 63;
            int r4  = (e >> 6) << 2;          // channel quad base 0..28
            int hw  = hw0 + col;
            float v0 = 0.f, v1 = 0.f, v2 = 0.f, v3 = 0.f;
            const float* xp = x + ((long)b * C + c0 + r4) * HWD + hw;
            if (hw < HWD) {
                v0 = xp[0]; v1 = xp[HWD]; v2 = xp[2 * HWD]; v3 = xp[3 * HWD];
                if (c0 + 32 < C) __builtin_prefetch(xp + 32 * HWD, 0, 0);
            }
            uint2 pv;
            pv.x = (unsigned)f2bfu(v0) | ((unsigned)f2bfu(v1) << 16);
            pv.y = (unsigned)f2bfu(v2) | ((unsigned)f2bfu(v3) << 16);
            *(uint2*)&lds_xT[col][r4] = pv;
        }
        __syncthreads();

        // ||x||^2: one thread per pixel column, vectorized LDS reads
        if (tid < 64) {
            const uint4* row = (const uint4*)&lds_xT[tid][0];
            #pragma unroll
            for (int q = 0; q < 4; ++q) {
                uint4 d = row[q];
                float p0 = bflo(d.x), p1 = bfhi(d.x), p2 = bflo(d.y), p3 = bfhi(d.y);
                float p4 = bflo(d.z), p5 = bfhi(d.z), p6 = bflo(d.w), p7 = bfhi(d.w);
                bsq += p0 * p0 + p1 * p1 + p2 * p2 + p3 * p3
                     + p4 * p4 + p5 * p5 + p6 * p6 + p7 * p7;
            }
        }

        if (tileValid) {  // wave-uniform predicate: EXEC all-ones inside
            const uint4* wp4 = (const uint4*)(wland_bf + lm * C + c0 + abase);
            U32x8s au; au.a = wp4[0]; au.b = wp4[2];      // K {abase..+7, abase+16..+23}
            v16bf afrag = __builtin_bit_cast(v16bf, au);

            const uint4* bp4 = (const uint4*)&lds_xT[lcol][rb];
            U32x8s bu; bu.a = bp4[0]; bu.b = bp4[2];      // rows {rb..+7, rb+16..+23}
            v16bf bfrag = __builtin_bit_cast(v16bf, bu);

            acc = __builtin_amdgcn_wmma_f32_16x16x32_bf16(
                false, afrag, false, bfrag, (short)0, acc, false, false);
        }
    }

    if (tileValid) {
        #pragma unroll
        for (int i = 0; i < 8; ++i) sm_ab[lcol][i + (lhalf ? 8 : 0)] = acc[i];
    }
    __syncthreads();

    if (tid < 64) {
        int hw = hw0 + tid;
        if (hw < HWD) {
            float d[K], mx = -3.4e38f;
            #pragma unroll
            for (int k = 0; k < K; ++k) {
                float dist = bsq - 2.0f * sm_ab[tid][k] + a_sq_g[k];
                d[k] = -dist;
                mx = fmaxf(mx, d[k]);
            }
            float s = 0.0f;
            #pragma unroll
            for (int k = 0; k < K; ++k) { d[k] = __expf(d[k] - mx); s += d[k]; }
            float inv = 1.0f / s;
            #pragma unroll
            for (int k = 0; k < K; ++k) {
                float m = d[k] * inv;
                maps_out[((long)b * K + k) * HWD + hw] = m;
                atomicAdd(&sattn[k], m);
            }
        }
    }
    __syncthreads();
    if (tid < K) atomicAdd(&attn_out[b * K + tid], sattn[tid]);
}

// ---------------- kernel 2: all_features = X * maps^T / HW, modulated ----------------
// grid = Bn*24, block = 256 (8 waves). The per-batch maps slab (K x HWD fp32,
// 34.5KB) is DMA'd once per block into LDS by the Tensor Data Mover, then all
// waves stream it with 16B LDS loads. x is read with unguarded b128 loads for
// the 24 full K-chunks; the 16-wide tail is handled explicitly.
__global__ __launch_bounds__(256)
void feat_kernel(const float* __restrict__ x,
                 const float* __restrict__ maps,
                 const float* __restrict__ modulation,
                 float* __restrict__ feat_out) {
    const int b    = blockIdx.x / 24;
    const int cblk = blockIdx.x % 24;
    const int tid  = threadIdx.x;
    const int wave = tid >> 5;
    const int lane = tid & 31;
    const int cBase = cblk * 128 + wave * 16;
    const int lm    = lane & 15;
    const int lhalf = lane >> 4;
    const int abase = lhalf ? 8 : 0;
    const int rb    = lhalf ? 8 : 0;

    __shared__ __align__(16) float lds_maps[K * HWD];

    // ---- TDM: stage maps[b] (2D tensor: dim0=HWD, dim1=K, 4B elems) into LDS ----
    if (tid < 32) {  // single wave issues the DMA (EXEC ignored by TENSOR ops)
        unsigned long long gaddr = (unsigned long long)(const void*)(maps + (long)b * K * HWD);
        unsigned lds_off = (unsigned)(unsigned long long)(uintptr_t)&lds_maps[0];
        // D# group0: count=1 | lds_addr | global_addr[56:0] | type=2
        u32x4 g0 = { 1u,
                     lds_off,
                     (unsigned)gaddr,
                     ((unsigned)(gaddr >> 32) & 0x01FFFFFFu) | (2u << 30) };
        // D# group1: data_size=4B; tensor_dim0=HWD; tensor_dim1=K;
        //            tile_dim0=HWD; tile_dim1=K; tensor_dim0_stride=HWD
        u32x8 g1 = { 2u << 16,
                     ((unsigned)HWD & 0xFFFFu) << 16,
                     ((unsigned)HWD >> 16) | (((unsigned)K & 0xFFFFu) << 16),
                     (((unsigned)K >> 16) & 0xFFFFu) | ((unsigned)HWD << 16),
                     (unsigned)K,
                     (unsigned)HWD,
                     0u, 0u };
        u32x4 gz = { 0u, 0u, 0u, 0u };  // groups 2/3: unused dims (tile_dim2..4 = 0)
        asm volatile("tensor_load_to_lds %0, %1, %2, %3"
                     :: "s"(g0), "s"(g1), "s"(gz), "s"(gz)
                     : "memory");
        __builtin_amdgcn_s_wait_tensorcnt(0);
    }
    __syncthreads();

    v8f acc = {};
    const long   xrow = ((long)b * C + (cBase + lm)) * HWD;
    const float* mrow = lds_maps + lm * HWD;  // meaningful only when lm < K

    for (int hw0 = 0; hw0 < 768; hw0 += 32) {   // 24 full, guard-free iterations
        const float4* xp = (const float4*)(x + xrow + hw0 + abase);
        float4 a0 = xp[0], a1 = xp[1], a2 = xp[4], a3 = xp[5];
        v16bf afrag;
        afrag[0]  = f2bf(a0.x); afrag[1]  = f2bf(a0.y); afrag[2]  = f2bf(a0.z); afrag[3]  = f2bf(a0.w);
        afrag[4]  = f2bf(a1.x); afrag[5]  = f2bf(a1.y); afrag[6]  = f2bf(a1.z); afrag[7]  = f2bf(a1.w);
        afrag[8]  = f2bf(a2.x); afrag[9]  = f2bf(a2.y); afrag[10] = f2bf(a2.z); afrag[11] = f2bf(a2.w);
        afrag[12] = f2bf(a3.x); afrag[13] = f2bf(a3.y); afrag[14] = f2bf(a3.z); afrag[15] = f2bf(a3.w);

        float4 m0 = {0,0,0,0}, m1 = {0,0,0,0}, m2 = {0,0,0,0}, m3 = {0,0,0,0};
        if (lm < K) {
            const float4* mp = (const float4*)(mrow + hw0 + rb);
            m0 = mp[0]; m1 = mp[1]; m2 = mp[4]; m3 = mp[5];
        }
        v16bf bfrag;
        bfrag[0]  = f2bf(m0.x); bfrag[1]  = f2bf(m0.y); bfrag[2]  = f2bf(m0.z); bfrag[3]  = f2bf(m0.w);
        bfrag[4]  = f2bf(m1.x); bfrag[5]  = f2bf(m1.y); bfrag[6]  = f2bf(m1.z); bfrag[7]  = f2bf(m1.w);
        bfrag[8]  = f2bf(m2.x); bfrag[9]  = f2bf(m2.y); bfrag[10] = f2bf(m2.z); bfrag[11] = f2bf(m2.w);
        bfrag[12] = f2bf(m3.x); bfrag[13] = f2bf(m3.y); bfrag[14] = f2bf(m3.z); bfrag[15] = f2bf(m3.w);

        acc = __builtin_amdgcn_wmma_f32_16x16x32_bf16(
            false, afrag, false, bfrag, (short)0, acc, false, false);
    }

    {   // tail: hw0 = 768, only hw 768..783 are valid (low 16 of the 32-chunk)
        const float4* xp = (const float4*)(x + xrow + 768 + abase);
        float4 a0 = xp[0], a1 = xp[1];
        __bf16 z = f2bf(0.0f);
        v16bf afrag;
        afrag[0]  = f2bf(a0.x); afrag[1]  = f2bf(a0.y); afrag[2]  = f2bf(a0.z); afrag[3]  = f2bf(a0.w);
        afrag[4]  = f2bf(a1.x); afrag[5]  = f2bf(a1.y); afrag[6]  = f2bf(a1.z); afrag[7]  = f2bf(a1.w);
        #pragma unroll
        for (int i = 8; i < 16; ++i) afrag[i] = z;   // K = hw0+abase+16.. is OOB

        float4 m0 = {0,0,0,0}, m1 = {0,0,0,0}, m2 = {0,0,0,0}, m3 = {0,0,0,0};
        if (lm < K) {                                // rows rb+16.. are OOB
            const float4* mp = (const float4*)(mrow + 768 + rb);
            m0 = mp[0]; m1 = mp[1];
        }
        v16bf bfrag;
        bfrag[0]  = f2bf(m0.x); bfrag[1]  = f2bf(m0.y); bfrag[2]  = f2bf(m0.z); bfrag[3]  = f2bf(m0.w);
        bfrag[4]  = f2bf(m1.x); bfrag[5]  = f2bf(m1.y); bfrag[6]  = f2bf(m1.z); bfrag[7]  = f2bf(m1.w);
        bfrag[8]  = f2bf(m2.x); bfrag[9]  = f2bf(m2.y); bfrag[10] = f2bf(m2.z); bfrag[11] = f2bf(m2.w);
        bfrag[12] = f2bf(m3.x); bfrag[13] = f2bf(m3.y); bfrag[14] = f2bf(m3.z); bfrag[15] = f2bf(m3.w);

        acc = __builtin_amdgcn_wmma_f32_16x16x32_bf16(
            false, afrag, false, bfrag, (short)0, acc, false, false);
    }

    const float invHW = 1.0f / (float)HWD;
    if (lm < K) {
        #pragma unroll
        for (int i = 0; i < 8; ++i) {
            int crow = cBase + i + (lhalf ? 8 : 0);
            float v = acc[i] * invHW * modulation[crow * K + lm];
            feat_out[((long)b * C + crow) * K + lm] = v;
        }
    }
}

// ---------------- kernel 3: g[b,c] = sum_k feat_mod[b,c,k] ----------------
__global__ void gsum_kernel(const float* __restrict__ feat, float* __restrict__ g) {
    int idx = blockIdx.x * blockDim.x + threadIdx.x;
    if (idx < Bn * C) {
        const float* p = feat + (long)idx * K;
        float s = 0.0f;
        #pragma unroll
        for (int k = 0; k < K; ++k) s += p[k];
        g[idx] = s;
    }
}

// ---------------- kernel 4: scores[b,n] = sum_c g[b,c] * w_cls[n,c] ----------------
__global__ __launch_bounds__(64)
void score_kernel(const float* __restrict__ g, const float* __restrict__ w_cls,
                  float* __restrict__ scores) {
    int b = blockIdx.x / NC;
    int n = blockIdx.x % NC;
    int tid = threadIdx.x;
    float s = 0.0f;
    for (int c = tid; c < C; c += 64) s += g[b * C + c] * w_cls[n * C + c];
    __shared__ float red[64];
    red[tid] = s;
    __syncthreads();
    for (int off = 32; off > 0; off >>= 1) {
        if (tid < off) red[tid] += red[tid + off];
        __syncthreads();
    }
    if (tid == 0) scores[b * NC + n] = red[0];
}

// ---------------- launcher ----------------
extern "C" void kernel_launch(void* const* d_in, const int* in_sizes, int n_in,
                              void* d_out, int out_size, void* d_ws, size_t ws_size,
                              hipStream_t stream) {
    const float* x          = (const float*)d_in[0];
    const float* w_land     = (const float*)d_in[1];
    const float* modulation = (const float*)d_in[2];
    const float* w_cls      = (const float*)d_in[3];

    float* out        = (float*)d_out;
    float* feat_out   = out;                            // (B,C,K)
    float* scores_out = feat_out + Bn * C * K;          // (B,NC)
    float* maps_out   = scores_out + Bn * NC;           // (B,K,HW)
    float* attn_out   = maps_out + (long)Bn * K * HWD;  // (B,K)

    // workspace: [ bf16 w_land (KP*C) | a_sq (16 f32) | g (B*C f32) ]
    __bf16* wland_bf = (__bf16*)d_ws;
    float*  a_sq     = (float*)((char*)d_ws + (size_t)KP * C * sizeof(__bf16));
    float*  gbuf     = a_sq + 16;

    init_kernel<<<64, 256, 0, stream>>>(w_land, attn_out, wland_bf, a_sq);
    maps_kernel<<<Bn * 13, 128, 0, stream>>>(x, wland_bf, a_sq, maps_out, attn_out);
    feat_kernel<<<Bn * 24, 256, 0, stream>>>(x, maps_out, modulation, feat_out);
    gsum_kernel<<<(Bn * C + 255) / 256, 256, 0, stream>>>(feat_out, gbuf);
    score_kernel<<<Bn * NC, 64, 0, stream>>>(gbuf, w_cls, scores_out);
}
